// EfficientSelfAttention_43026982371635
// MI455X (gfx1250) — matrix operand order repaired and checked by reading, not compiled, not run on hardware
//
#include <hip/hip_runtime.h>
#include <hip/hip_bf16.h>
#include <math.h>

// ---------------------------------------------------------------------------
// EfficientSelfAttention for MI455X (gfx1250), wave32 + WMMA bf16.
// Sizes fixed by the reference: B=4, T=4096, D=1024, H=8, Dh=128.
// HBM-bound workload: all GEMM operands staged bf16; fp32 accumulate.
// q-softmax fused into GEMM epilogue (block N-tile == one head); k kept bf16.
// ---------------------------------------------------------------------------

#define BQ   4
#define TT   4096
#define DD   1024
#define HH   8
#define DH   128
#define NROW (BQ * TT)      // 16384 rows
#define KSPLIT 8            // T-split for the K^T V GEMM
#define NEGM (-1000000.0f)

typedef __bf16 bf16_t;
typedef __attribute__((ext_vector_type(16))) __bf16 v16bf;
typedef __attribute__((ext_vector_type(8)))  __bf16 v8bf;
typedef __attribute__((ext_vector_type(8)))  float  v8f;

// ---------------------------------------------------------------------------
// WMMA helpers (CDNA5 16x16x32 bf16 -> f32). Layouts per CDNA5 ISA.
// ---------------------------------------------------------------------------

__device__ __forceinline__ v8f v8f_zero() {
    v8f z;
#pragma unroll
    for (int i = 0; i < 8; ++i) z[i] = 0.0f;
    return z;
}

// A fragment: 16x32 (MxK) bf16 tile, row stride `stride` halfs.
// lanes 0-15 : M=lane,   halves 0..7 = K 0..7,  halves 8..15 = K 16..23
// lanes 16-31: M=lane-16,halves 0..7 = K 8..15, halves 8..15 = K 24..31
__device__ __forceinline__ v16bf frag_load_a(const bf16_t* base, int stride) {
    int lane = threadIdx.x & 31;
    int m    = lane & 15;
    int kb   = (lane & 16) >> 1;      // 0 or 8
    const bf16_t* r = base + m * stride + kb;
    union { v16bf v; v8bf h[2]; } u;
    u.h[0] = *(const v8bf*)(r);
    u.h[1] = *(const v8bf*)(r + 16);
    return u.v;
}

// B fragment from a transposed tile Bt[n][k] (row stride `stride` halfs).
// lanes 0-15: N=lane, halves 0..7 = K 0..7, 8..15 = K 16..23; lanes 16-31: N=lane-16, K 8..15, 24..31.
__device__ __forceinline__ v16bf frag_load_b(const bf16_t* base, int stride) {
    int lane = threadIdx.x & 31;
    int n    = lane & 15;
    int kb   = (lane & 16) >> 1;      // 0 or 8
    const bf16_t* r = base + n * stride + kb;
    union { v16bf v; v8bf h[2]; } u;
    u.h[0] = *(const v8bf*)(r);
    u.h[1] = *(const v8bf*)(r + 16);
    return u.v;
}

__device__ __forceinline__ v8f wmma_bf16(v16bf a, v16bf b, v8f c) {
    return __builtin_amdgcn_wmma_f32_16x16x32_bf16(
        /*neg_a=*/false, a, /*neg_b=*/false, b,
        /*c_mod=*/(short)0, c, /*reuse_a=*/false, /*reuse_b=*/false);
}

// ---------------------------------------------------------------------------
// 1) Weight fp32 -> bf16 with transpose:  Wt[n*D + k] = (bf16) W[k*D + n]
// ---------------------------------------------------------------------------
__global__ __launch_bounds__(256) void wcvt_kernel(const float* __restrict__ W,
                                                   bf16_t* __restrict__ Wt) {
    int idx = blockIdx.x * 256 + threadIdx.x;   // over D*D, output-major
    int n = idx >> 10;
    int k = idx & 1023;
    Wt[idx] = (bf16_t)W[(size_t)k * DD + n];
}

// ---------------------------------------------------------------------------
// 2) LayerNorm per row of 1024, write bf16 xn.
// ---------------------------------------------------------------------------
__global__ __launch_bounds__(256) void ln_kernel(const float* __restrict__ x,
                                                 const float* __restrict__ gamma,
                                                 const float* __restrict__ beta,
                                                 bf16_t* __restrict__ xnb) {
    int row = blockIdx.x;
    int tid = threadIdx.x;
    const float* xr = x + (size_t)row * DD;
    float v[4], s = 0.f, sq = 0.f;
#pragma unroll
    for (int i = 0; i < 4; ++i) {
        v[i] = xr[tid + 256 * i];
        s  += v[i];
        sq += v[i] * v[i];
    }
    __shared__ float rs[256], rq[256];
    rs[tid] = s; rq[tid] = sq;
    __syncthreads();
    for (int off = 128; off > 0; off >>= 1) {
        if (tid < off) { rs[tid] += rs[tid + off]; rq[tid] += rq[tid + off]; }
        __syncthreads();
    }
    float mu  = rs[0] * (1.0f / DD);
    float var = rq[0] * (1.0f / DD) - mu * mu;
    float inv = rsqrtf(var + 1e-5f);
#pragma unroll
    for (int i = 0; i < 4; ++i) {
        int c = tid + 256 * i;
        float xn = (v[i] - mu) * inv * gamma[c] + beta[c];
        xnb[(size_t)row * DD + c] = (bf16_t)xn;
    }
}

// ---------------------------------------------------------------------------
// 3) QKV GEMM:  C[16384x1024] = xn_bf16 @ W  (+bias epilogue), bf16 output.
//    Block tile 64x128 (N-tile == one head), BK=32; 8 waves (2x4),
//    wave tile 32x32 (4 WMMA accums).
//    mode 0: q -> fused per-row softmax over the 128-wide head, bf16 out
//    mode 1: k -> + (1-mask)*NEG, bf16 out (pre-softmax values)
//    mode 2: v -> * mask, bf16 out
// ---------------------------------------------------------------------------
__global__ __launch_bounds__(256) void gemm_qkv_kernel(
        const bf16_t* __restrict__ A, const bf16_t* __restrict__ Bt,
        const float* __restrict__ bias, const float* __restrict__ mask,
        bf16_t* __restrict__ Cb, int mode) {
    __shared__ alignas(16) bf16_t As[64 * 32];    // [m][k]
    __shared__ alignas(16) bf16_t Bs[128 * 32];   // [n][k] (transposed)
    __shared__ alignas(16) float  Cs[64 * 128];   // fp32 tile for q softmax
    int tid = threadIdx.x;
    int m0  = blockIdx.y * 64;
    int n0  = blockIdx.x * 128;
    int w   = tid >> 5;
    int wm  = w >> 2, wn = w & 3;

    v8f acc[2][2];
#pragma unroll
    for (int mi = 0; mi < 2; ++mi)
#pragma unroll
        for (int ni = 0; ni < 2; ++ni) acc[mi][ni] = v8f_zero();

    int ar = tid >> 2, aseg = tid & 3;            // A stage: 64 rows x 4 x 16B
    int bn = tid >> 1, bseg = tid & 1;            // B stage: 128 rows x 2 x 32B

    for (int step = 0; step < 32; ++step) {
        int k0 = step * 32;
        *(v8bf*)&As[ar * 32 + aseg * 8] =
            *(const v8bf*)&A[(size_t)(m0 + ar) * DD + k0 + aseg * 8];
        *(v8bf*)&Bs[bn * 32 + bseg * 16] =
            *(const v8bf*)&Bt[(size_t)(n0 + bn) * DD + k0 + bseg * 16];
        *(v8bf*)&Bs[bn * 32 + bseg * 16 + 8] =
            *(const v8bf*)&Bt[(size_t)(n0 + bn) * DD + k0 + bseg * 16 + 8];
        if (step + 1 < 32) {   // gfx1250 global_prefetch of next tiles
            __builtin_prefetch(&A[(size_t)(m0 + ar) * DD + k0 + 32 + aseg * 8], 0, 3);
            __builtin_prefetch(&Bt[(size_t)(n0 + bn) * DD + k0 + 32 + bseg * 16], 0, 3);
        }
        __syncthreads();
        v16bf af[2], bfg[2];
#pragma unroll
        for (int mi = 0; mi < 2; ++mi)
            af[mi] = frag_load_a(&As[(wm * 32 + mi * 16) * 32], 32);
#pragma unroll
        for (int ni = 0; ni < 2; ++ni)
            bfg[ni] = frag_load_b(&Bs[(wn * 32 + ni * 16) * 32], 32);
#pragma unroll
        for (int mi = 0; mi < 2; ++mi)
#pragma unroll
            for (int ni = 0; ni < 2; ++ni)
                acc[mi][ni] = wmma_bf16(af[mi], bfg[ni], acc[mi][ni]);
        __syncthreads();
    }

    int lane  = tid & 31;
    int cn    = lane & 15;
    int rbase = (lane & 16) >> 1;                 // 0 or 8

    if (mode == 0) {
        // q: park fp32 tile in LDS, then per-row softmax (row == one head).
#pragma unroll
        for (int mi = 0; mi < 2; ++mi)
#pragma unroll
            for (int ni = 0; ni < 2; ++ni)
#pragma unroll
                for (int r = 0; r < 8; ++r) {
                    int rr  = wm * 32 + mi * 16 + rbase + r;
                    int col = wn * 32 + ni * 16 + cn;
                    Cs[rr * 128 + col] = acc[mi][ni][r] + bias[n0 + col];
                }
        __syncthreads();
        // each wave owns 8 rows; 4 elements per lane per row
#pragma unroll
        for (int rr2 = 0; rr2 < 8; ++rr2) {
            int r = w * 8 + rr2;
            float v[4];
#pragma unroll
            for (int i = 0; i < 4; ++i) v[i] = Cs[r * 128 + lane * 4 + i];
            float m = fmaxf(fmaxf(v[0], v[1]), fmaxf(v[2], v[3]));
#pragma unroll
            for (int off = 16; off > 0; off >>= 1)
                m = fmaxf(m, __shfl_xor(m, off, 32));
            float s = 0.f;
#pragma unroll
            for (int i = 0; i < 4; ++i) { v[i] = expf(v[i] - m); s += v[i]; }
#pragma unroll
            for (int off = 16; off > 0; off >>= 1) s += __shfl_xor(s, off, 32);
            float inv = 1.0f / s;
#pragma unroll
            for (int i = 0; i < 4; ++i)
                Cb[(size_t)(m0 + r) * DD + n0 + lane * 4 + i] =
                    (bf16_t)(v[i] * inv);
        }
    } else {
#pragma unroll
        for (int mi = 0; mi < 2; ++mi)
#pragma unroll
            for (int ni = 0; ni < 2; ++ni)
#pragma unroll
                for (int r = 0; r < 8; ++r) {
                    int row = m0 + wm * 32 + mi * 16 + rbase + r;
                    int col = n0 + wn * 32 + ni * 16 + cn;
                    float val = acc[mi][ni][r] + bias[col];
                    if (mode == 1) val += (1.0f - mask[row]) * NEGM;
                    else           val *= mask[row];
                    Cb[(size_t)row * DD + col] = (bf16_t)val;
                }
    }
}

// ---------------------------------------------------------------------------
// 4a) k column softmax stats over bf16 pre-softmax k, split over T.
//     part[split][b*D + c][2] = online (max, sum)
// ---------------------------------------------------------------------------
__global__ __launch_bounds__(256) void kstats1_kernel(const bf16_t* __restrict__ kp,
                                                      float* __restrict__ part) {
    int col = blockIdx.x * 256 + threadIdx.x;     // 0..B*D-1
    int b   = col >> 10;
    int c   = col & 1023;
    int t0  = blockIdx.y * (TT / KSPLIT);
    float m = -INFINITY, s = 0.f;
    for (int t = t0; t < t0 + TT / KSPLIT; ++t) {
        float v = (float)kp[((size_t)(b * TT + t) << 10) + c];
        if (v > m) { s = s * expf(m - v) + 1.0f; m = v; }
        else       { s += expf(v - m); }
    }
    size_t o = ((size_t)blockIdx.y * (BQ * DD) + col) * 2;
    part[o + 0] = m;
    part[o + 1] = s;
}

// 4b) combine partials across splits.
__global__ __launch_bounds__(256) void kcombine_kernel(const float* __restrict__ part,
                                                       float* __restrict__ stats) {
    int col = blockIdx.x * 256 + threadIdx.x;
    float m = -INFINITY, s = 0.f;
#pragma unroll
    for (int sp = 0; sp < KSPLIT; ++sp) {
        size_t o = ((size_t)sp * (BQ * DD) + col) * 2;
        float mp = part[o], sq = part[o + 1];
        if (mp > m) { s = s * expf(m - mp) + sq; m = mp; }
        else        { s += sq * expf(mp - m); }
    }
    stats[col * 2 + 0] = m;
    stats[col * 2 + 1] = s;
}

// 4c) normalize k -> bf16.
__global__ __launch_bounds__(256) void knorm_kernel(const bf16_t* __restrict__ kp,
                                                    const float* __restrict__ stats,
                                                    bf16_t* __restrict__ kb) {
    size_t idx = (size_t)blockIdx.x * 256 + threadIdx.x;
    int c = (int)(idx & 1023);
    int b = (int)(idx >> 22);                     // T*D = 2^22
    float m = stats[(b * DD + c) * 2 + 0];
    float s = stats[(b * DD + c) * 2 + 1];
    kb[idx] = (bf16_t)(expf((float)kp[idx] - m) / s);
}

// ---------------------------------------------------------------------------
// 5) zero att accumulator.
// ---------------------------------------------------------------------------
__global__ __launch_bounds__(256) void zero_kernel(float* __restrict__ p) {
    p[(size_t)blockIdx.x * 256 + threadIdx.x] = 0.0f;
}

// ---------------------------------------------------------------------------
// 6) att[bh][d][l] += sum_t k[t,d] * v[t,l]  — WMMA, T split over grid.y,
//    fp32 atomic-add epilogue. One block covers the full 128x128 output.
//    8 waves as 4(M) x 2(N); wave tile 32x64 (8 accums).
// ---------------------------------------------------------------------------
__global__ __launch_bounds__(256) void att_gemm_kernel(const bf16_t* __restrict__ kb,
                                                       const bf16_t* __restrict__ vb,
                                                       float* __restrict__ attf) {
    __shared__ alignas(16) bf16_t As[128 * 32];   // [d][t_local]
    __shared__ alignas(16) bf16_t Bs[128 * 32];   // [l][t_local]
    int tid = threadIdx.x;
    int bh  = blockIdx.x;
    int b   = bh >> 3, h = bh & 7;
    int t0  = blockIdx.y * (TT / KSPLIT);
    int w   = tid >> 5;
    int wm  = w >> 1, wn = w & 1;

    v8f acc[2][4];
#pragma unroll
    for (int mi = 0; mi < 2; ++mi)
#pragma unroll
        for (int ni = 0; ni < 4; ++ni) acc[mi][ni] = v8f_zero();

    int srow = tid >> 3;   // 0..31 local t
    int sseg = tid & 7;    // 16-channel chunk

    for (int step = 0; step < (TT / KSPLIT) / 32; ++step) {
        int t = t0 + step * 32 + srow;
        size_t gb = ((size_t)(b * TT + t) << 10) + h * DH + sseg * 16;
        v8bf ka0 = *(const v8bf*)&kb[gb];
        v8bf ka1 = *(const v8bf*)&kb[gb + 8];
        v8bf va0 = *(const v8bf*)&vb[gb];
        v8bf va1 = *(const v8bf*)&vb[gb + 8];
#pragma unroll
        for (int i = 0; i < 8; ++i) {             // transpose into LDS
            As[(sseg * 16 + i) * 32 + srow]     = ka0[i];
            As[(sseg * 16 + 8 + i) * 32 + srow] = ka1[i];
            Bs[(sseg * 16 + i) * 32 + srow]     = va0[i];
            Bs[(sseg * 16 + 8 + i) * 32 + srow] = va1[i];
        }
        __syncthreads();
        v16bf af[2], bfg[4];
#pragma unroll
        for (int mi = 0; mi < 2; ++mi)
            af[mi] = frag_load_a(&As[(wm * 32 + mi * 16) * 32], 32);
#pragma unroll
        for (int ni = 0; ni < 4; ++ni)
            bfg[ni] = frag_load_b(&Bs[(wn * 64 + ni * 16) * 32], 32);
#pragma unroll
        for (int mi = 0; mi < 2; ++mi)
#pragma unroll
            for (int ni = 0; ni < 4; ++ni)
                acc[mi][ni] = wmma_bf16(af[mi], bfg[ni], acc[mi][ni]);
        __syncthreads();
    }

    int lane  = tid & 31;
    int cn    = lane & 15;
    int rbase = (lane & 16) >> 1;
#pragma unroll
    for (int mi = 0; mi < 2; ++mi)
#pragma unroll
        for (int ni = 0; ni < 4; ++ni)
#pragma unroll
            for (int r = 0; r < 8; ++r) {
                int d = wm * 32 + mi * 16 + rbase + r;
                int l = wn * 64 + ni * 16 + cn;
                atomicAdd(&attf[(size_t)bh * DH * DH + d * DH + l],
                          acc[mi][ni][r]);
            }
}

// ---------------------------------------------------------------------------
// 7) att fp32 -> bf16 with transpose: attb[bh][l][d] = attf[bh][d][l]
// ---------------------------------------------------------------------------
__global__ __launch_bounds__(256) void attcvt_kernel(const float* __restrict__ attf,
                                                     bf16_t* __restrict__ attb) {
    int bh = blockIdx.x;
    for (int j = 0; j < 64; ++j) {
        int i = threadIdx.x + j * 256;            // 0..16383
        int d = i >> 7, l = i & 127;
        attb[(size_t)bh * (DH * DH) + l * DH + d] =
            (bf16_t)attf[(size_t)bh * (DH * DH) + d * DH + l];
    }
}

// ---------------------------------------------------------------------------
// 8) y = q @ att per (b,h); block tile 64(t) x 128(l), K = 128 in 4 steps.
//    att tile (attb, [l][d]) fully resident in LDS. Epilogue: out = x + y.
// ---------------------------------------------------------------------------
__global__ __launch_bounds__(256) void y_gemm_kernel(const bf16_t* __restrict__ qb,
                                                     const bf16_t* __restrict__ attb,
                                                     const float* __restrict__ x,
                                                     float* __restrict__ out) {
    __shared__ alignas(16) bf16_t Bs[128 * 128];  // [l][d]
    __shared__ alignas(16) bf16_t As[64 * 32];    // [t][d-chunk]
    int tid = threadIdx.x;
    int bh  = blockIdx.y;
    int b   = bh >> 3, h = bh & 7;
    int tt0 = blockIdx.x * 64;

    {   // stage whole att tile: 128 rows x 128 halfs
        int l = tid >> 1, seg = tid & 1;
        const bf16_t* src = &attb[(size_t)bh * (DH * DH) + l * DH + seg * 64];
        bf16_t* dst = &Bs[l * DH + seg * 64];
#pragma unroll
        for (int i = 0; i < 8; ++i)
            *(v8bf*)&dst[i * 8] = *(const v8bf*)&src[i * 8];
    }

    int w  = tid >> 5;
    int wm = w >> 2, wn = w & 3;
    v8f acc[2][2];
#pragma unroll
    for (int mi = 0; mi < 2; ++mi)
#pragma unroll
        for (int ni = 0; ni < 2; ++ni) acc[mi][ni] = v8f_zero();

    int ar = tid >> 2, aseg = tid & 3;
    __syncthreads();

    for (int step = 0; step < 4; ++step) {
        int k0 = step * 32;
        *(v8bf*)&As[ar * 32 + aseg * 8] =
            *(const v8bf*)&qb[((size_t)(b * TT + tt0 + ar) << 10) + h * DH + k0 + aseg * 8];
        __syncthreads();
        v16bf af[2], bfg[2];
#pragma unroll
        for (int mi = 0; mi < 2; ++mi)
            af[mi] = frag_load_a(&As[(wm * 32 + mi * 16) * 32], 32);
#pragma unroll
        for (int ni = 0; ni < 2; ++ni)
            bfg[ni] = frag_load_b(&Bs[(wn * 32 + ni * 16) * DH + k0], DH);
#pragma unroll
        for (int mi = 0; mi < 2; ++mi)
#pragma unroll
            for (int ni = 0; ni < 2; ++ni)
                acc[mi][ni] = wmma_bf16(af[mi], bfg[ni], acc[mi][ni]);
        __syncthreads();
    }

    int lane  = tid & 31;
    int cn    = lane & 15;
    int rbase = (lane & 16) >> 1;
#pragma unroll
    for (int mi = 0; mi < 2; ++mi)
#pragma unroll
        for (int ni = 0; ni < 2; ++ni)
#pragma unroll
            for (int r = 0; r < 8; ++r) {
                int trow = tt0 + wm * 32 + mi * 16 + rbase + r;
                int col  = h * DH + wn * 32 + ni * 16 + cn;
                size_t gi = ((size_t)(b * TT + trow) << 10) + col;
                out[gi] = x[gi] + acc[mi][ni][r];
            }
}

// ---------------------------------------------------------------------------
// Host launcher
// ---------------------------------------------------------------------------
extern "C" void kernel_launch(void* const* d_in, const int* in_sizes, int n_in,
                              void* d_out, int out_size, void* d_ws, size_t ws_size,
                              hipStream_t stream) {
    const float* x     = (const float*)d_in[0];
    const float* mask  = (const float*)d_in[1];   // [B*T]
    const float* Wq    = (const float*)d_in[2];
    const float* bq    = (const float*)d_in[3];
    const float* Wk    = (const float*)d_in[4];
    const float* bk    = (const float*)d_in[5];
    const float* Wv    = (const float*)d_in[6];
    const float* bv    = (const float*)d_in[7];
    const float* gamma = (const float*)d_in[8];
    const float* beta  = (const float*)d_in[9];
    float* out = (float*)d_out;

    char* ws = (char*)d_ws;
    size_t o = 0;
    auto alloc = [&](size_t bytes) -> void* {
        void* p = ws + o;
        o += (bytes + 255) & ~(size_t)255;
        return p;
    };
    bf16_t* xnb  = (bf16_t*)alloc((size_t)NROW * DD * 2);       // 32 MB
    bf16_t* wqt  = (bf16_t*)alloc((size_t)DD * DD * 2);         // 2 MB
    bf16_t* wkt  = (bf16_t*)alloc((size_t)DD * DD * 2);
    bf16_t* wvt  = (bf16_t*)alloc((size_t)DD * DD * 2);
    bf16_t* qbb  = (bf16_t*)alloc((size_t)NROW * DD * 2);       // q (softmaxed)
    bf16_t* kpre = (bf16_t*)alloc((size_t)NROW * DD * 2);       // k pre-softmax
    bf16_t* kbb  = (bf16_t*)alloc((size_t)NROW * DD * 2);       // k (softmaxed)
    bf16_t* vbb  = (bf16_t*)alloc((size_t)NROW * DD * 2);       // v (masked)
    float*  attf = (float*) alloc((size_t)BQ * HH * DH * DH * 4);
    bf16_t* attb = (bf16_t*)alloc((size_t)BQ * HH * DH * DH * 2);
    float*  part = (float*) alloc((size_t)KSPLIT * BQ * DD * 2 * 4);
    float*  stat = (float*) alloc((size_t)BQ * DD * 2 * 4);
    (void)ws_size; (void)in_sizes; (void)n_in; (void)out_size;

    // 1) weights -> bf16 transposed
    wcvt_kernel<<<DD * DD / 256, 256, 0, stream>>>(Wq, wqt);
    wcvt_kernel<<<DD * DD / 256, 256, 0, stream>>>(Wk, wkt);
    wcvt_kernel<<<DD * DD / 256, 256, 0, stream>>>(Wv, wvt);

    // 2) LayerNorm
    ln_kernel<<<NROW, 256, 0, stream>>>(x, gamma, beta, xnb);

    // 3) QKV GEMMs (WMMA bf16); q softmax fused in epilogue
    dim3 ggrid(DD / 128, NROW / 64);
    gemm_qkv_kernel<<<ggrid, 256, 0, stream>>>(xnb, wqt, bq, mask, qbb, 0);
    gemm_qkv_kernel<<<ggrid, 256, 0, stream>>>(xnb, wkt, bk, mask, kpre, 1);
    gemm_qkv_kernel<<<ggrid, 256, 0, stream>>>(xnb, wvt, bv, mask, vbb, 2);

    // 4) k softmax (time dim), bf16 in/out with fp32 stats
    kstats1_kernel<<<dim3(BQ * DD / 256, KSPLIT), 256, 0, stream>>>(kpre, part);
    kcombine_kernel<<<BQ * DD / 256, 256, 0, stream>>>(part, stat);
    knorm_kernel<<<(size_t)NROW * DD / 256, 256, 0, stream>>>(kpre, stat, kbb);

    // 5-7) att = K^T V (WMMA + f32 atomics), then bf16 transpose
    zero_kernel<<<BQ * HH * DH * DH / 256, 256, 0, stream>>>(attf);
    att_gemm_kernel<<<dim3(BQ * HH, KSPLIT), 256, 0, stream>>>(kbb, vbb, attf);
    attcvt_kernel<<<BQ * HH, 256, 0, stream>>>(attf, attb);

    // 8) y = q @ att, fused residual add
    y_gemm_kernel<<<dim3(TT / 64, BQ * HH), 256, 0, stream>>>(qbb, attb, x, out);
}
